// Conv_23038204576396
// MI455X (gfx1250) — matrix operand order repaired and checked by reading, not compiled, not run on hardware
//
#include <hip/hip_runtime.h>
#include <cstdint>
#include <cstddef>

// ---------------------------------------------------------------------------
// Sparse (Minkowski) convolution for MI455X (gfx1250, wave32, WMMA bf16).
//   - W[k] (128x256) staged ONCE per WG into LDS as bf16, transposed+padded.
//   - Each WG processes 5 m-tiles of 32 gather/scatter pairs for its k.
//   - Gather is software-pipelined: next tile's rows load into VGPRs while
//     the current tile's v_wmma_f32_16x16x32_bf16 ops run.
//   - f32 accumulate, global_atomic_add_f32 scatter.
// ---------------------------------------------------------------------------

typedef __attribute__((ext_vector_type(16))) __bf16 v16bf;
typedef __attribute__((ext_vector_type(8)))  __bf16 v8bf;
typedef __attribute__((ext_vector_type(8)))  float  v8f;

#define K_OFF        27
#define M_PAIRS      20000
#define C_IN         128
#define C_OUT        256
#define TILE_M       32
#define GX           125          // grid.x : 125 WG-columns
#define TILES_PER_WG 5            // 125 * 5 = 625 = M_PAIRS / TILE_M
#define A_STRIDE     136          // bf16 elems per A row (128 + 8 pad, 16B-mult)
#define B_STRIDE     136          // bf16 elems per B^T row (128 + 8 pad)

union BF16Frag {
    v16bf v;
    struct { v8bf lo, hi; } h;
};

__global__ __launch_bounds__(256)
void spconv_zero_kernel(float4* __restrict__ p, int n4)
{
    int i = blockIdx.x * blockDim.x + threadIdx.x;
    if (i < n4) p[i] = make_float4(0.f, 0.f, 0.f, 0.f);
}

__global__ __launch_bounds__(256)
void spconv_wmma_kernel(const float* __restrict__ x,            // [N_IN, C_IN]
                        const float* __restrict__ Wg,           // [K, C_IN, C_OUT]
                        const long long* __restrict__ in_idx,   // [K, M]
                        const long long* __restrict__ out_idx,  // [K, M]
                        float* __restrict__ out)                // [N_OUT, C_OUT]
{
    __shared__ __align__(16) __bf16 lds_B[C_OUT * B_STRIDE];   // 69632 B : W[k]^T bf16
    __shared__ __align__(16) __bf16 lds_A[TILE_M * A_STRIDE];  //  8704 B : gathered rows
    __shared__ int lds_orow[TILE_M];

    const int tid = threadIdx.x;
    const int k   = blockIdx.y;                // kernel offset 0..26
    const int r   = tid >> 3;                  // gather: row of tile (0..31)
    const int cb  = (tid & 7) * 16;            // gather: 16-channel block
    const size_t idx_base = (size_t)k * M_PAIRS;

    int tile = blockIdx.x;                     // m-tile id, strided by GX

    // ---- prefetch tile 0: gathered rows (f32) + output row index ----
    float4 pre[4];
    {
        const long long gi = in_idx[idx_base + (size_t)tile * TILE_M + r];
        const float4* src = (const float4*)(x + (size_t)gi * C_IN + cb);
        pre[0] = src[0]; pre[1] = src[1]; pre[2] = src[2]; pre[3] = src[3];
    }
    int orow_next = 0;
    if (tid < TILE_M)
        orow_next = (int)out_idx[idx_base + (size_t)tile * TILE_M + tid];

    // ---- stage full W[k] transposed into LDS as bf16: thread == column n ----
    {
        const float* wp = Wg + (size_t)k * C_IN * C_OUT + tid;   // column n = tid
        __bf16* bt = &lds_B[tid * B_STRIDE];
        #pragma unroll 8
        for (int c = 0; c < C_IN; ++c)
            bt[c] = (__bf16)wp[(size_t)c * C_OUT];
    }

    // wave tiling: 8 waves = 2 (M halves of 16) x 4 (N strips of 64)
    const int wave   = tid >> 5;
    const int lane   = tid & 31;
    const int laneHi = lane >> 4;
    const int m0     = (wave & 1) * 16;
    const int n0     = (wave >> 1) * 64;
    const int arow   = m0 + (lane & 15);
    const int ncol   = lane & 15;

    #pragma unroll 1
    for (int it = 0; it < TILES_PER_WG; ++it) {
        // ---- commit prefetched A tile to LDS (f32 -> bf16, native cvt) ----
        {
            __bf16* dst = &lds_A[r * A_STRIDE + cb];
            #pragma unroll
            for (int i = 0; i < 4; ++i) {
                dst[i*4 + 0] = (__bf16)pre[i].x;
                dst[i*4 + 1] = (__bf16)pre[i].y;
                dst[i*4 + 2] = (__bf16)pre[i].z;
                dst[i*4 + 3] = (__bf16)pre[i].w;
            }
        }
        if (tid < TILE_M) lds_orow[tid] = orow_next;
        __syncthreads();   // A tile + (first iter) W tile + orow visible

        // ---- prefetch next tile while this tile computes ----
        if (it + 1 < TILES_PER_WG) {
            const int ntile = tile + GX;
            const long long gi = in_idx[idx_base + (size_t)ntile * TILE_M + r];
            const float4* src = (const float4*)(x + (size_t)gi * C_IN + cb);
            pre[0] = src[0]; pre[1] = src[1]; pre[2] = src[2]; pre[3] = src[3];
            if (tid < TILE_M)
                orow_next = (int)out_idx[idx_base + (size_t)ntile * TILE_M + tid];
        }

        // ---- GEMM: 16 x v_wmma_f32_16x16x32_bf16 per wave ----
        v8f acc0 = {}, acc1 = {}, acc2 = {}, acc3 = {};
        #pragma unroll
        for (int kc = 0; kc < 4; ++kc) {
            BF16Frag a;   // A 16x32: lanes 0-15 K 0..7/16..23, lanes 16-31 +8
            const __bf16* ap = &lds_A[arow * A_STRIDE + kc * 32 + laneHi * 8];
            a.h.lo = *(const v8bf*)(ap);
            a.h.hi = *(const v8bf*)(ap + 16);
            #pragma unroll
            for (int nt = 0; nt < 4; ++nt) {
                BF16Frag b;  // B 32x16: lanes 0-15 K 0..7/16..23, lanes 16-31 +8
                const __bf16* bp =
                    &lds_B[(n0 + nt * 16 + ncol) * B_STRIDE + kc * 32 + laneHi * 8];
                b.h.lo = *(const v8bf*)(bp);
                b.h.hi = *(const v8bf*)(bp + 16);
                v8f c = (nt == 0) ? acc0 : (nt == 1) ? acc1 : (nt == 2) ? acc2 : acc3;
                c = __builtin_amdgcn_wmma_f32_16x16x32_bf16(
                        false, a.v, false, b.v, (short)0, c, false, false);
                if      (nt == 0) acc0 = c;
                else if (nt == 1) acc1 = c;
                else if (nt == 2) acc2 = c;
                else              acc3 = c;
            }
        }

        // ---- scatter-add: VGPR rr, lane L -> M = rr + 8*(L>=16), N = L&15 ----
        #pragma unroll
        for (int rr = 0; rr < 8; ++rr) {
            const int m = m0 + laneHi * 8 + rr;
            float* op = out + (size_t)lds_orow[m] * C_OUT + n0 + ncol;
            atomicAdd(op + 0,  acc0[rr]);
            atomicAdd(op + 16, acc1[rr]);
            atomicAdd(op + 32, acc2[rr]);
            atomicAdd(op + 48, acc3[rr]);
        }
        __syncthreads();   // everyone done with lds_A/lds_orow before overwrite
        tile += GX;
    }
}

extern "C" void kernel_launch(void* const* d_in, const int* in_sizes, int n_in,
                              void* d_out, int out_size, void* d_ws, size_t ws_size,
                              hipStream_t stream) {
    const float*     x       = (const float*)d_in[0];      // [200000,128] f32
    const float*     Wg      = (const float*)d_in[1];      // [27,128,256] f32
    const long long* in_idx  = (const long long*)d_in[2];  // [27,20000] i64
    const long long* out_idx = (const long long*)d_in[3];  // [27,20000] i64
    // d_in[4] = n_out scalar (known: 100000) -- unused
    float* out = (float*)d_out;

    // zero the (poisoned) output buffer
    {
        int n4 = out_size / 4;
        int blocks = (n4 + 255) / 256;
        spconv_zero_kernel<<<blocks, 256, 0, stream>>>((float4*)out, n4);
    }

    dim3 grid(GX, K_OFF);   // 125 x 27 WGs, 5 m-tiles each
    spconv_wmma_kernel<<<grid, 256, 0, stream>>>(x, Wg, in_idx, out_idx, out);
}
